// QuantizedLinear158_53987738911497
// MI455X (gfx1250) — compile-verified, matrix operand order checked by reading, not run on hardware
//
#include <hip/hip_runtime.h>
#include <hip/hip_bf16.h>

// ---- problem constants (from reference) ----
#define IN_F   1024
#define OUT_F  1024
#define NTOK   32768               // B*S = 8*4096
#define NWORDS (OUT_F * IN_F / 4)  // 262144 packed ternary words

typedef int v8i __attribute__((ext_vector_type(8)));
typedef int v2i __attribute__((ext_vector_type(2)));

// -------------------------------------------------------------------------
// Pass 1: per-token int8 fake-quant of activations.
// One 256-thread block per token; each thread handles 4 contiguous floats.
// q8:   [NTOK][IN_F] int8
// invs: [NTOK] float  (= max|x| / 127 = 1/act_scale)
// -------------------------------------------------------------------------
__global__ __launch_bounds__(256)
void ql158_act_quant(const float* __restrict__ x,
                     signed char* __restrict__ q8,
                     float* __restrict__ invs) {
    const int token = blockIdx.x;
    const int t     = threadIdx.x;                 // 0..255
    const float4 v  = ((const float4*)(x + (size_t)token * IN_F))[t];

    float m = fmaxf(fmaxf(fabsf(v.x), fabsf(v.y)),
                    fmaxf(fabsf(v.z), fabsf(v.w)));

    // wave32 reduction
    #pragma unroll
    for (int off = 16; off > 0; off >>= 1)
        m = fmaxf(m, __shfl_xor(m, off, 32));

    __shared__ float red[8];
    if ((t & 31) == 0) red[t >> 5] = m;
    __syncthreads();

    float bm = red[0];
    #pragma unroll
    for (int i = 1; i < 8; ++i) bm = fmaxf(bm, red[i]);

    const float mx = fmaxf(bm, 1e-5f);
    const float s  = 127.0f / mx;                  // act_scale

    // round (RNE, matches jnp.round) then clip, like the reference
    float r0 = fminf(fmaxf(rintf(v.x * s), -128.f), 127.f);
    float r1 = fminf(fmaxf(rintf(v.y * s), -128.f), 127.f);
    float r2 = fminf(fmaxf(rintf(v.z * s), -128.f), 127.f);
    float r3 = fminf(fmaxf(rintf(v.w * s), -128.f), 127.f);

    int packed = ((int)r0 & 255) | (((int)r1 & 255) << 8) |
                 (((int)r2 & 255) << 16) | (((int)r3 & 255) << 24);
    ((int*)(q8 + (size_t)token * IN_F))[t] = packed;

    if (t == 0) invs[token] = mx * (1.0f / 127.0f);  // 1/act_scale
}

// -------------------------------------------------------------------------
// Pass 2: unpack ternary weights, reproducing the reference's bit-plane
// permutation:  w[o,i] = ((packed[(o&255)*1024 + i] >> (2*(o>>8))) & 3) - 1
// w8: [OUT_F][IN_F] int8 (row-major by output feature o)
// -------------------------------------------------------------------------
__global__ __launch_bounds__(256)
void ql158_unpack_w(const int* __restrict__ pw,
                    signed char* __restrict__ w8) {
    const int wi = blockIdx.x * 256 + threadIdx.x;  // 0..NWORDS-1
    const int word  = pw[wi];
    const int i     = wi & (IN_F - 1);
    const int obase = wi >> 10;                     // 0..255
    #pragma unroll
    for (int p = 0; p < 4; ++p) {
        const int o = (p << 8) | obase;
        w8[(size_t)o * IN_F + i] =
            (signed char)(((word >> (2 * p)) & 3) - 1);
    }
}

// -------------------------------------------------------------------------
// Pass 3: int8 GEMM via V_WMMA_I32_16X16X64_IU8.
// Grid (NTOK/64, OUT_F/128); block = 256 threads = 8 waves.
// Wave (mw = w&3, nw = w>>2): 16(M) x 64(N) tile, 4 accumulators.
// 8-bit operand layout (ISA 7.12.2): lane = {lo=row/col, h=lane>>4};
// VGPR 2p,2p+1 hold K = kb + 16p + 8h .. +7  (one b64 load each pair).
// -------------------------------------------------------------------------
__global__ __launch_bounds__(256)
void ql158_gemm_iu8(const signed char* __restrict__ q8,   // [NTOK][IN_F]
                    const signed char* __restrict__ w8,   // [OUT_F][IN_F]
                    const float* __restrict__ invs,       // [NTOK]
                    const float* __restrict__ bias,       // [OUT_F]
                    const float* __restrict__ wscale_p,   // [1]
                    float* __restrict__ out) {            // [NTOK][OUT_F]
    const int lane = threadIdx.x & 31;
    const int wv   = threadIdx.x >> 5;         // 0..7
    const int lo   = lane & 15;
    const int h    = lane >> 4;

    const int mbase = blockIdx.x * 64  + (wv & 3) * 16;
    const int nbase = blockIdx.y * 128 + (wv >> 2) * 64;

    const signed char* arow = q8 + (size_t)(mbase + lo) * IN_F;
    const signed char* bc0  = w8 + (size_t)(nbase +  0 + lo) * IN_F;
    const signed char* bc1  = w8 + (size_t)(nbase + 16 + lo) * IN_F;
    const signed char* bc2  = w8 + (size_t)(nbase + 32 + lo) * IN_F;
    const signed char* bc3  = w8 + (size_t)(nbase + 48 + lo) * IN_F;

    // distinct accumulator chains (helps regalloc keep WMMA dest == phi regs)
    v8i acc0 = {}, acc1 = {}, acc2 = {}, acc3 = {};

    #pragma unroll 2
    for (int kb = 0; kb < IN_F; kb += 64) {
        // prefetch next A chunk into near caches (WGP-scope)
        __builtin_prefetch(arow + kb + 64, 0, 3);

        // ---- A fragment: 16x64 int8, row = mbase+lo ----
        v8i a;
        #pragma unroll
        for (int p = 0; p < 4; ++p) {
            v2i t = *(const v2i*)(arow + kb + 16 * p + 8 * h);
            a[2 * p]     = t.x;
            a[2 * p + 1] = t.y;
        }

        v8i b0, b1, b2, b3;
        #pragma unroll
        for (int p = 0; p < 4; ++p) {
            v2i u = *(const v2i*)(bc0 + kb + 16 * p + 8 * h);
            b0[2 * p] = u.x; b0[2 * p + 1] = u.y;
        }
        #pragma unroll
        for (int p = 0; p < 4; ++p) {
            v2i u = *(const v2i*)(bc1 + kb + 16 * p + 8 * h);
            b1[2 * p] = u.x; b1[2 * p + 1] = u.y;
        }
        #pragma unroll
        for (int p = 0; p < 4; ++p) {
            v2i u = *(const v2i*)(bc2 + kb + 16 * p + 8 * h);
            b2[2 * p] = u.x; b2[2 * p + 1] = u.y;
        }
        #pragma unroll
        for (int p = 0; p < 4; ++p) {
            v2i u = *(const v2i*)(bc3 + kb + 16 * p + 8 * h);
            b3[2 * p] = u.x; b3[2 * p + 1] = u.y;
        }

        // signed A (int8 acts), signed B (ternary weights)
        acc0 = __builtin_amdgcn_wmma_i32_16x16x64_iu8(true, a, true, b0, acc0, false, false);
        acc1 = __builtin_amdgcn_wmma_i32_16x16x64_iu8(true, a, true, b1, acc1, false, false);
        acc2 = __builtin_amdgcn_wmma_i32_16x16x64_iu8(true, a, true, b2, acc2, false, false);
        acc3 = __builtin_amdgcn_wmma_i32_16x16x64_iu8(true, a, true, b3, acc3, false, false);
    }

    // ---- epilogue: D layout lane->N (lo), VGPR v -> M = v + 8h ----
    const float wsc = wscale_p[0];
    float rs[8];
    #pragma unroll
    for (int v = 0; v < 8; ++v)
        rs[v] = invs[mbase + 8 * h + v] * wsc;

    const v8i accs[4] = {acc0, acc1, acc2, acc3};
    #pragma unroll
    for (int t = 0; t < 4; ++t) {
        const int col = nbase + t * 16 + lo;
        const float bs = bias[col];
        #pragma unroll
        for (int v = 0; v < 8; ++v) {
            const int mrow = mbase + 8 * h + v;
            out[(size_t)mrow * OUT_F + col] =
                (float)accs[t][v] * rs[v] + bs;
        }
    }
}

// -------------------------------------------------------------------------
extern "C" void kernel_launch(void* const* d_in, const int* in_sizes, int n_in,
                              void* d_out, int out_size, void* d_ws, size_t ws_size,
                              hipStream_t stream) {
    const float* x      = (const float*)d_in[0];   // 33554432 f32
    const int*   pw     = (const int*)d_in[1];     // 262144 int32
    const float* wscale = (const float*)d_in[2];   // 1 f32
    const float* bias   = (const float*)d_in[3];   // 1024 f32
    float*       out    = (float*)d_out;           // 33554432 f32

    char* ws = (char*)d_ws;
    signed char* q8   = (signed char*)ws;                              // 32 MiB
    float*       invs = (float*)(ws + (size_t)NTOK * IN_F);            // 128 KiB
    signed char* w8   = (signed char*)(ws + (size_t)NTOK * IN_F
                                          + (size_t)NTOK * 4);         // 1 MiB

    ql158_act_quant<<<NTOK, 256, 0, stream>>>(x, q8, invs);
    ql158_unpack_w<<<NWORDS / 256, 256, 0, stream>>>(pw, w8);
    ql158_gemm_iu8<<<dim3(NTOK / 64, OUT_F / 128, 1), 256, 0, stream>>>(
        q8, w8, invs, bias, wscale, out);
}
